// SoftmaxMatcher_20366734917819
// MI455X (gfx1250) — matrix operand order repaired and checked by reading, not compiled, not run on hardware
//
#include <hip/hip_runtime.h>

typedef __attribute__((ext_vector_type(16))) _Float16 v16h;
typedef __attribute__((ext_vector_type(8)))  _Float16 v8h;
typedef __attribute__((ext_vector_type(8)))  float    v8f;
typedef __attribute__((ext_vector_type(4)))  unsigned int u32x4;
typedef __attribute__((ext_vector_type(8)))  unsigned int u32x8;

#define BP 8      // pairs
#define C  128    // descriptor channels
#define NN 4096   // keypoints
#define CV 144    // 128 desc + x + y + weight + 13 pad  (9 tiles of 16)
#define TM 64     // m-tile (softmax/GEMM column tile)
#define PROW 80   // P scratch row pitch in halves (16B aligned per lane)
#define LOG2E 1.44269504088896340736f

// ---------------- prep: normalize + pack to f16 layouts ----------------
__global__ void prep_kernel(const float* __restrict__ coords,
                            const float* __restrict__ wts,
                            const float* __restrict__ desc,
                            _Float16* __restrict__ Qn,
                            _Float16* __restrict__ Kn,
                            _Float16* __restrict__ Vt) {
  int p = blockIdx.y;
  int n = blockIdx.x * blockDim.x + threadIdx.x;
  if (n >= NN) return;

  const float* sd = desc + (size_t)(2 * p) * C * NN;
  float ss = 0.f;
  for (int c = 0; c < C; ++c) { float v = sd[(size_t)c * NN + n]; ss += v * v; }
  float sinv = 1.f / fmaxf(sqrtf(ss), 1e-12f);
  _Float16* q = Qn + ((size_t)p * NN + n) * C;
  for (int c = 0; c < C; ++c) q[c] = (_Float16)(sd[(size_t)c * NN + n] * sinv);

  const float* td = desc + (size_t)(2 * p + 1) * C * NN;
  float ts = 0.f;
  for (int c = 0; c < C; ++c) { float v = td[(size_t)c * NN + n]; ts += v * v; }
  float tinv = 1.f / fmaxf(sqrtf(ts), 1e-12f);
  _Float16* k  = Kn + ((size_t)p * NN + n) * C;
  _Float16* vt = Vt + (size_t)p * CV * NN;
  for (int c = 0; c < C; ++c) {
    float v = td[(size_t)c * NN + n];
    k[c] = (_Float16)(v * tinv);
    vt[(size_t)c * NN + n] = (_Float16)v;
  }
  vt[(size_t)128 * NN + n] = (_Float16)coords[((size_t)(2 * p + 1) * 2 + 0) * NN + n];
  vt[(size_t)129 * NN + n] = (_Float16)coords[((size_t)(2 * p + 1) * 2 + 1) * NN + n];
  vt[(size_t)130 * NN + n] = (_Float16)wts[(size_t)(2 * p + 1) * NN + n];
  for (int c = 131; c < CV; ++c) vt[(size_t)c * NN + n] = (_Float16)0.f;
}

// K tile: contiguous 16KB -> per-lane async copies to LDS (ASYNCcnt)
__device__ __forceinline__ void stage_k_async(const _Float16* kbase, int m0,
                                              unsigned lds_base, int tid) {
  unsigned gbase = (unsigned)m0 * (C * 2);
#pragma unroll
  for (int j = 0; j < 4; ++j) {
    unsigned i = (unsigned)tid + (unsigned)j * 256u;
    unsigned loff = lds_base + i * 16u;
    unsigned goff = gbase + i * 16u;
    asm volatile("global_load_async_to_lds_b128 %0, %1, %2"
                 :: "v"(loff), "v"(goff), "s"(kbase) : "memory");
  }
}

// V tile: 2D strided tile (CV rows x TM halves, row stride NN) -> TDM (TENSORcnt)
__device__ __forceinline__ void stage_v_tdm(const _Float16* vbase, int m0,
                                            unsigned lds_addr) {
  unsigned long long ga = (unsigned long long)(size_t)(vbase + m0);
  u32x4 g0;
  g0.x = 1u;                                   // count=1, user mode, no gather
  g0.y = lds_addr;                             // LDS byte address
  g0.z = (unsigned)ga;                         // global_addr[31:0]
  g0.w = ((unsigned)(ga >> 32) & 0x01FFFFFFu) | (2u << 30);  // addr[56:32] | type=2
  u32x8 g1;
  g1[0] = 1u << 16;                            // wg_mask=0, data_size=2B
  g1[1] = ((unsigned)NN & 0xFFFFu) << 16;      // tensor_dim0 lo16 (bits 63:48)
  g1[2] = ((unsigned)NN >> 16) | ((unsigned)CV << 16);  // td0 hi16 | tensor_dim1 lo16
  g1[3] = (unsigned)TM << 16;                  // td1 hi16(=0) | tile_dim0=64
  g1[4] = (unsigned)CV;                        // tile_dim1=144, tile_dim2=0
  g1[5] = (unsigned)NN;                        // tensor_dim0_stride lo32
  g1[6] = 0u;                                  // stride0 hi | stride1 lo
  g1[7] = 0u;                                  // stride1 hi
  u32x4 gz = {0u, 0u, 0u, 0u};                 // groups 2/3: dims unused (2D tile)
  asm volatile("tensor_load_to_lds %0, %1, %2, %3"
               :: "s"(g0), "s"(g1), "s"(gz), "s"(gz) : "memory");
}

// ---------------- fused flash-style matcher ----------------
__global__ __launch_bounds__(256)
void matcher_kernel(const float* __restrict__ coords,
                    const float* __restrict__ wts,
                    const float* __restrict__ desc,
                    const _Float16* __restrict__ Qn,
                    const _Float16* __restrict__ Kn,
                    const _Float16* __restrict__ Vt,
                    float* __restrict__ out) {
  __shared__ _Float16 Ksh[2][TM * C];        // [buf][m][c]
  __shared__ _Float16 Vsh[2][CV * TM];       // [buf][cv][m]
  __shared__ _Float16 Psh[8 * 16 * PROW];    // per-wave [row][m + pad]

  const int tid  = threadIdx.x;
  const int wave = tid >> 5;
  const int lane = tid & 31;
  const int li   = lane & 15;
  const int hi   = lane >> 4;
  const int p    = blockIdx.y;
  const int n0   = blockIdx.x * 128 + wave * 16;

  const _Float16* kbase = Kn + (size_t)p * NN * C;
  const _Float16* vbase = Vt + (size_t)p * CV * NN;
  const unsigned klds[2] = {(unsigned)(size_t)&Ksh[0][0], (unsigned)(size_t)&Ksh[1][0]};
  const unsigned vlds[2] = {(unsigned)(size_t)&Vsh[0][0], (unsigned)(size_t)&Vsh[1][0]};

  // Q A-fragments: contiguous k order, each lane's fragment is the
  // 16-half run [kc*32+16*hi, +16), loaded as two 8-half pieces.
  v16h qa[4];
  {
    const _Float16* qb = Qn + ((size_t)p * NN + n0 + li) * C;
#pragma unroll
    for (int kc = 0; kc < 4; ++kc) {
      v8h lo = *(const v8h*)&qb[kc * 32 + 16 * hi];
      v8h hh = *(const v8h*)&qb[kc * 32 + 16 * hi + 8];
#pragma unroll
      for (int e = 0; e < 8; ++e) { qa[kc][e] = lo[e]; qa[kc][e + 8] = hh[e]; }
    }
  }

  v8f acc[9];
#pragma unroll
  for (int t = 0; t < 9; ++t) acc[t] = (v8f){};
  float m_run[8], l_run[8];
#pragma unroll
  for (int r = 0; r < 8; ++r) { m_run[r] = -3.0e38f; l_run[r] = 0.f; }

  // prologue: stage tile 0 into buffer 0
  stage_k_async(kbase, 0, klds[0], tid);
  if (wave == 0) stage_v_tdm(vbase, 0, vlds[0]);

  const int NT = NN / TM;
  for (int mi = 0; mi < NT; ++mi) {
    const int buf = mi & 1;
    if (wave == 0) __builtin_amdgcn_s_wait_tensorcnt(0);
    asm volatile("s_wait_asynccnt 0" ::: "memory");
    __syncthreads();  // buffer `buf` is ready for everyone
    if (mi + 1 < NT) {  // kick off next tile into the other buffer
      stage_k_async(kbase, (mi + 1) * TM, klds[buf ^ 1], tid);
      if (wave == 0) stage_v_tdm(vbase, (mi + 1) * TM, vlds[buf ^ 1]);
    }

    // S = Q * K^T : 16x64 per wave (4 subtiles x 4 k-chunks = 16 WMMAs)
    v8f s[4];
#pragma unroll
    for (int ms = 0; ms < 4; ++ms) {
      v8f cacc = (v8f){};
#pragma unroll
      for (int kc = 0; kc < 4; ++kc) {
        v16h b = *(const v16h*)&Ksh[buf][(ms * 16 + li) * C + kc * 32 + hi * 16];
        cacc = __builtin_amdgcn_wmma_f32_16x16x32_f16(false, qa[kc], false, b,
                                                      (short)0, cacc, false, false);
      }
      s[ms] = cacc;
    }

    // online softmax; C-layout: element r = row r (lanes0-15) / r+8 (lanes16-31)
    _Float16* pv = Psh + wave * (16 * PROW);
#pragma unroll
    for (int r = 0; r < 8; ++r) {
      float z0 = s[0][r] * 100.f, z1 = s[1][r] * 100.f;
      float z2 = s[2][r] * 100.f, z3 = s[3][r] * 100.f;
      float mx = fmaxf(fmaxf(z0, z1), fmaxf(z2, z3));
      mx = fmaxf(mx, __shfl_xor(mx, 1));
      mx = fmaxf(mx, __shfl_xor(mx, 2));
      mx = fmaxf(mx, __shfl_xor(mx, 4));
      mx = fmaxf(mx, __shfl_xor(mx, 8));
      float mnew  = fmaxf(m_run[r], mx);
      float alpha = exp2f((m_run[r] - mnew) * LOG2E);
      float p0 = exp2f((z0 - mnew) * LOG2E), p1 = exp2f((z1 - mnew) * LOG2E);
      float p2 = exp2f((z2 - mnew) * LOG2E), p3 = exp2f((z3 - mnew) * LOG2E);
      float ps = p0 + p1 + p2 + p3;
      ps += __shfl_xor(ps, 1); ps += __shfl_xor(ps, 2);
      ps += __shfl_xor(ps, 4); ps += __shfl_xor(ps, 8);
      l_run[r] = l_run[r] * alpha + ps;
      m_run[r] = mnew;
#pragma unroll
      for (int t = 0; t < 9; ++t) acc[t][r] *= alpha;
      int row = r + 8 * hi;
      pv[row * PROW +  0 + li] = (_Float16)p0;
      pv[row * PROW + 16 + li] = (_Float16)p1;
      pv[row * PROW + 32 + li] = (_Float16)p2;
      pv[row * PROW + 48 + li] = (_Float16)p3;
    }
    asm volatile("s_wait_dscnt 0" ::: "memory");  // same-wave LDS RAW fence

    // O += P * V^T : 2 k-chunks x 9 cv-tiles = 18 WMMAs
#pragma unroll
    for (int kc = 0; kc < 2; ++kc) {
      const _Float16* pr = pv + li * PROW;
      v8h lo = *(const v8h*)&pr[kc * 32 + 16 * hi];
      v8h hh = *(const v8h*)&pr[kc * 32 + 16 * hi + 8];
      v16h a;
#pragma unroll
      for (int e = 0; e < 8; ++e) { a[e] = lo[e]; a[e + 8] = hh[e]; }
#pragma unroll
      for (int t = 0; t < 9; ++t) {
        v16h b = *(const v16h*)&Vsh[buf][(t * 16 + li) * TM + kc * 32 + hi * 16];
        acc[t] = __builtin_amdgcn_wmma_f32_16x16x32_f16(false, a, false, b,
                                                        (short)0, acc[t], false, false);
      }
    }
  }

  // epilogue: divide by denom, renormalize desc, score vs raw src, write outputs
  const float* sd = desc + (size_t)(2 * p) * C * NN;
#pragma unroll
  for (int r = 0; r < 8; ++r) {
    float invl = 1.f / fmaxf(l_run[r], 1e-37f);
    int row = r + 8 * hi;
    int n = n0 + row;
    float s2 = 0.f;
#pragma unroll
    for (int t = 0; t < 8; ++t) { float pd = acc[t][r] * invl; s2 += pd * pd; }
    s2 += __shfl_xor(s2, 1); s2 += __shfl_xor(s2, 2);
    s2 += __shfl_xor(s2, 4); s2 += __shfl_xor(s2, 8);
    float innv = 1.f / fmaxf(sqrtf(s2), 1e-12f);
    float sc = 0.f;
#pragma unroll
    for (int t = 0; t < 8; ++t) {
      int c = t * 16 + li;
      sc += sd[(size_t)c * NN + n] * (acc[t][r] * invl);
    }
    sc += __shfl_xor(sc, 1); sc += __shfl_xor(sc, 2);
    sc += __shfl_xor(sc, 4); sc += __shfl_xor(sc, 8);
    float score = sc * innv * (1.f / 128.f);
    float ov = acc[8][r] * invl;  // lane li: 0=x, 1=y, 2=pseudo_weight
    if (li < 2) out[((size_t)p * 2 + li) * NN + n] = ov;
    if (li == 2) {
      float sw = wts[(size_t)(2 * p) * NN + n];
      out[(size_t)BP * 2 * NN + (size_t)p * NN + n] = 0.5f * (score + 1.f) * sw * ov;
    }
  }
}

extern "C" void kernel_launch(void* const* d_in, const int* in_sizes, int n_in,
                              void* d_out, int out_size, void* d_ws, size_t ws_size,
                              hipStream_t stream) {
  const float* coords = (const float*)d_in[0];
  const float* wts    = (const float*)d_in[1];
  const float* desc   = (const float*)d_in[2];
  float* out = (float*)d_out;

  _Float16* Qn = (_Float16*)d_ws;
  _Float16* Kn = Qn + (size_t)BP * NN * C;
  _Float16* Vt = Kn + (size_t)BP * NN * C;

  dim3 pb(256), pg(NN / 256, BP);
  prep_kernel<<<pg, pb, 0, stream>>>(coords, wts, desc, Qn, Kn, Vt);

  dim3 mb(256), mg(NN / 128, BP);
  matcher_kernel<<<mg, mb, 0, stream>>>(coords, wts, desc, Qn, Kn, Vt, out);
}
